// DualLaplacianBlock_40149354283526
// MI455X (gfx1250) — matrix operand order repaired and checked by reading, not compiled, not run on hardware
//
#include <hip/hip_runtime.h>
#include <math.h>
#include <stdint.h>

// ---------------- problem constants ----------------
#define BB 2
#define NN 4096
#define DD 256
#define T_WAKE 0.1f
#define W_L (1.0f - T_WAKE)
#define W_G (T_WAKE)
#define EPS 1e-8f

typedef __bf16 bf16;
typedef __attribute__((ext_vector_type(16))) bf16  v16bf;
typedef __attribute__((ext_vector_type(8)))  float v8f;

// ---------------- WMMA helper ----------------
// D = A(16x32 bf16) * B(32x16 bf16) + C(16x16 f32)
static __device__ __forceinline__ v8f wmma_bf16(v16bf a, v16bf b, v8f c) {
  return __builtin_amdgcn_wmma_f32_16x16x32_bf16(
      /*neg_a=*/false, a, /*neg_b=*/false, b,
      /*c_mod=*/(short)0, c, /*reuse_a=*/false, /*reuse_b=*/false);
}

// Build a 16-element bf16 fragment from two 16B chunks (8 bf16 each).
static __device__ __forceinline__ v16bf frag2x8(const bf16* p0, const bf16* p1) {
  union { v16bf v; uint4 q[2]; } u;
  u.q[0] = *(const uint4*)p0;
  u.q[1] = *(const uint4*)p1;
  return u.v;
}

// A-matrix fragment (16xK tile, row-major src). Per ISA 16-bit A layout:
// lanes 0-15 row=lane: K chunks {k0..k0+7, k0+16..k0+23}
// lanes 16-31          K chunks {k0+8..k0+15, k0+24..k0+31}
static __device__ __forceinline__ v16bf loadA_bf16(const bf16* src, int row,
                                                   int ld, int k0, int lane) {
  const bf16* p = src + (size_t)row * ld + k0 + ((lane & 16) ? 8 : 0);
  return frag2x8(p, p + 16);
}

// B-matrix fragment: B[k][n] = X[n][k] with X row-major (ld = K stride).
// lanes 0-15: n=lane, k = {k0..k0+7, k0+16..k0+23} ; lanes 16-31: n=lane-16, k = {k0+8..k0+15, k0+24..k0+31}
static __device__ __forceinline__ v16bf loadB_bf16(const bf16* src, int nrow,
                                                   int ld, int k0, int lane) {
  const bf16* p = src + (size_t)nrow * ld + k0 + ((lane & 16) ? 8 : 0);
  return frag2x8(p, p + 16);
}

// A fragment sourced from fp32, converted to bf16 on the fly.
static __device__ __forceinline__ v16bf loadA_f32(const float* src, int row,
                                                  int ld, int k0, int lane) {
  const float* p = src + (size_t)row * ld + k0 + ((lane & 16) ? 8 : 0);
  v16bf f;
#pragma unroll
  for (int i = 0; i < 8; ++i) {
    f[i]     = (bf16)p[i];
    f[8 + i] = (bf16)p[i + 16];
  }
  return f;
}

// ---------------- K0: transpose+convert a DxD weight to bf16 ----------------
__global__ __launch_bounds__(256) void k_transpose_w(const float* __restrict__ W,
                                                     bf16* __restrict__ WT) {
  int i = blockIdx.x;   // row of W
  int j = threadIdx.x;  // col of W
  WT[(size_t)j * DD + i] = (bf16)W[(size_t)i * DD + j];
}

// ---------------- K1: Y[M,256] = X[M,256] (f32) @ W, with WT bf16 ----------
__global__ __launch_bounds__(256) void k_gemm(const float* __restrict__ X,
                                              const bf16* __restrict__ WT,
                                              float* __restrict__ Y) {
  const int lane = threadIdx.x & 31;
  const int wv   = threadIdx.x >> 5;
  const int rowbase = blockIdx.x * 16;
  const int colbase = (blockIdx.y * 8 + wv) * 16;
  const int l15 = lane & 15;
  const int hi8 = (lane & 16) ? 8 : 0;

  v8f acc = {};
#pragma unroll
  for (int k0 = 0; k0 < DD; k0 += 32) {
    v16bf af = loadA_f32(X, rowbase + l15, DD, k0, lane);
    v16bf bfrag = loadB_bf16(WT, colbase + l15, DD, k0, lane);
    acc = wmma_bf16(af, bfrag, acc);
  }
#pragma unroll
  for (int j = 0; j < 8; ++j)
    Y[(size_t)(rowbase + j + hi8) * DD + colbase + l15] = acc[j];
}

// ---------------- K2: per-row prep -----------------------------------------
__global__ __launch_bounds__(256) void k_prep(const float* __restrict__ zl,
                                              const float* __restrict__ zg,
                                              const float* __restrict__ v,
                                              bf16* __restrict__ znl,
                                              bf16* __restrict__ zgb,
                                              bf16* __restrict__ vT,
                                              float* __restrict__ sq) {
  const int row = blockIdx.x;          // b*N + n
  const int b = row / NN, n = row % NN;
  const int d = threadIdx.x;
  const size_t idx = (size_t)row * DD + d;

  float a = zl[idx], g = zg[idx], vv = v[idx];
  __shared__ float r1[256], r2[256];
  r1[d] = a * a;
  r2[d] = g * g;
  __syncthreads();
#pragma unroll
  for (int s = 128; s > 0; s >>= 1) {
    if (d < s) { r1[d] += r1[d + s]; r2[d] += r2[d + s]; }
    __syncthreads();
  }
  float nrm = fmaxf(sqrtf(r1[0]), EPS);
  znl[idx] = (bf16)(a / nrm);
  zgb[idx] = (bf16)g;
  vT[((size_t)b * DD + d) * NN + n] = (bf16)vv;
  if (d == 0) sq[row] = r2[0];
}

// ---------------- K3: fused dual-Laplacian flash pass -----------------------
// u[b,n,:] = w_l * (K_l @ v) + w_g * (K_g @ v), causal, row-stochastic.
// A *pair* of waves shares one 16-row query tile; each wave handles half the
// key range. Degrees combined via LDS after pass 1; partial 16x256 output
// accumulators combined chunk-by-chunk via LDS after pass 2.
__global__ __launch_bounds__(256, 1) void k_flash(const bf16* __restrict__ znl,
                                                  const bf16* __restrict__ zgb,
                                                  const bf16* __restrict__ vT,
                                                  const float* __restrict__ sq,
                                                  float* __restrict__ u) {
  const int lane = threadIdx.x & 31;
  const int wv   = threadIdx.x >> 5;        // 0..7
  const int pair = wv >> 1;                 // 0..3
  const int half = wv & 1;                  // key-range half
  const int qg   = blockIdx.x * 4 + pair;   // 0 .. B*N/16-1
  const int b    = qg >> 8;                 // / (N/16)
  const int q    = qg & 255;
  const int rowbase = q * 16;

  const bf16* znlB = znl + (size_t)b * NN * DD;
  const bf16* zgB  = zgb + (size_t)b * NN * DD;
  const bf16* vTB  = vT + (size_t)b * DD * NN;
  const float* sqB = sq + (size_t)b * NN;

  __shared__ __align__(16) bf16 ldsT[8][16 * 32];      // per-wave T staging (8KB)
  __shared__ float denLDS[8][2][16];                   // per-wave partial degrees
  __shared__ __align__(16) float accLDS[4][16 * 16];   // per-pair combine chunk

  bf16* tls = ldsT[wv];
  const int l15 = lane & 15;
  const int hiR = (lane & 16) ? 8 : 0;

  int   rn[8];
  float sqq[8];
#pragma unroll
  for (int j = 0; j < 8; ++j) {
    rn[j]  = rowbase + j + hiR;
    sqq[j] = sqB[rn[j]];
  }

  // Hoisted query-side A fragments (loop-invariant over the whole key range).
  v16bf qal[8], qag[8];
#pragma unroll
  for (int kk = 0; kk < 8; ++kk) {
    qal[kk] = loadA_bf16(znlB, rowbase + l15, DD, kk * 32, lane);
    qag[kk] = loadA_bf16(zgB,  rowbase + l15, DD, kk * 32, lane);
  }

  // ---------------- pass 1: partial degrees over this wave's key half -------
  const int nt = q + 1;                       // 16-key tiles needed
  const int t0 = half ? (nt >> 1) : 0;
  const int t1 = half ? nt : (nt >> 1);
  v8f denl = {}, deng = {};
  for (int t = t0; t < t1; ++t) {
    const int mb = t * 16;
    if (t + 1 < t1) {  // stream next key tile (global_prefetch_b8)
      __builtin_prefetch(znlB + (size_t)(mb + 16 + l15) * DD, 0, 3);
      __builtin_prefetch(zgB  + (size_t)(mb + 16 + l15) * DD, 0, 3);
    }
    v8f sl = {}, sg = {};
#pragma unroll
    for (int kk = 0; kk < 8; ++kk) {
      sl = wmma_bf16(qal[kk], loadB_bf16(znlB, mb + l15, DD, kk * 32, lane), sl);
      sg = wmma_bf16(qag[kk], loadB_bf16(zgB,  mb + l15, DD, kk * 32, lane), sg);
    }
    const int m = mb + l15;
    const float sqm = sqB[m];
#pragma unroll
    for (int j = 0; j < 8; ++j) {
      const bool keep = m < rn[j];
      float vl = keep ? fmaxf(sl[j], 0.0f) : 0.0f;
      float d2 = fmaxf(sqq[j] + sqm - 2.0f * sg[j], 0.0f);
      float vg = keep ? __expf(-0.5f * d2) : 0.0f;
      denl[j] += vl;
      deng[j] += vg;
    }
  }

  // reduce rowsums across the 16-lane group, publish partial degrees
#pragma unroll
  for (int j = 0; j < 8; ++j) {
    float a = denl[j], g = deng[j];
#pragma unroll
    for (int off = 8; off > 0; off >>= 1) {
      a += __shfl_xor(a, off, 16);
      g += __shfl_xor(g, off, 16);
    }
    if (l15 == 0) {                      // lanes 0 and 16 cover rows 0..15
      denLDS[wv][0][j + hiR] = a;
      denLDS[wv][1][j + hiR] = g;
    }
  }
  __syncthreads();

  float coefl[8], coefg[8];
  {
    const int w0 = pair * 2, w1 = pair * 2 + 1;
#pragma unroll
    for (int j = 0; j < 8; ++j) {
      float a = denLDS[w0][0][j + hiR] + denLDS[w1][0][j + hiR];
      float g = denLDS[w0][1][j + hiR] + denLDS[w1][1][j + hiR];
      coefl[j] = W_L / fmaxf(a, EPS);
      coefg[j] = W_G / fmaxf(g, EPS);
    }
  }

  // ---------------- pass 2: partial u = T @ v over this wave's key half -----
  v8f acc[16];
  {
    v8f z = {};
#pragma unroll
    for (int dc = 0; dc < 16; ++dc) acc[dc] = z;
  }

  const int ns = (rowbase + 16 + 31) >> 5;    // 32-key steps
  const int s0 = half ? (ns >> 1) : 0;
  const int s1 = half ? ns : (ns >> 1);
  for (int s = s0; s < s1; ++s) {
    const int mb32 = s * 32;
    if (s + 1 < s1) {
      __builtin_prefetch(znlB + (size_t)(mb32 + 32 + l15) * DD, 0, 3);
      __builtin_prefetch(zgB  + (size_t)(mb32 + 32 + l15) * DD, 0, 3);
    }
#pragma unroll
    for (int sub = 0; sub < 2; ++sub) {
      const int mb = mb32 + sub * 16;
      v8f sl = {}, sg = {};
#pragma unroll
      for (int kk = 0; kk < 8; ++kk) {
        sl = wmma_bf16(qal[kk], loadB_bf16(znlB, mb + l15, DD, kk * 32, lane), sl);
        sg = wmma_bf16(qag[kk], loadB_bf16(zgB,  mb + l15, DD, kk * 32, lane), sg);
      }
      const int m = mb + l15;
      const float sqm = sqB[m];
#pragma unroll
      for (int j = 0; j < 8; ++j) {
        const bool keep = m < rn[j];
        float vl = keep ? fmaxf(sl[j], 0.0f) : 0.0f;
        float d2 = fmaxf(sqq[j] + sqm - 2.0f * sg[j], 0.0f);
        float vg = keep ? __expf(-0.5f * d2) : 0.0f;
        float tval = coefl[j] * vl + coefg[j] * vg;
        tls[(size_t)(j + hiR) * 32 + sub * 16 + l15] = (bf16)tval;
      }
    }
    // same-wave LDS RAW fence (C-layout stores -> A-fragment loads)
    asm volatile("s_wait_dscnt 0x0" ::: "memory");
    v16bf tf = frag2x8(tls + (size_t)l15 * 32 + ((lane & 16) ? 8 : 0),
                       tls + (size_t)l15 * 32 + ((lane & 16) ? 8 : 0) + 16);
#pragma unroll
    for (int dc = 0; dc < 16; ++dc) {
      v16bf bfrag = loadB_bf16(vTB, dc * 16 + l15, NN, mb32, lane);
      acc[dc] = wmma_bf16(tf, bfrag, acc[dc]);
    }
  }

  // ---------------- combine wave-pair partials, write u ---------------------
  // Fixed 16-iteration barrier loop: every wave executes the same barriers.
#pragma unroll 1
  for (int dc = 0; dc < 16; ++dc) {
    if (half == 1) {
#pragma unroll
      for (int j = 0; j < 8; ++j)
        accLDS[pair][(j + hiR) * 16 + l15] = acc[dc][j];
    }
    __syncthreads();
    if (half == 0) {
#pragma unroll
      for (int j = 0; j < 8; ++j) {
        float r = acc[dc][j] + accLDS[pair][(j + hiR) * 16 + l15];
        u[((size_t)b * NN + rn[j]) * DD + dc * 16 + l15] = r;
      }
    }
    __syncthreads();
  }
}

// ---------------- host-side launch ----------------
extern "C" void kernel_launch(void* const* d_in, const int* in_sizes, int n_in,
                              void* d_out, int out_size, void* d_ws, size_t ws_size,
                              hipStream_t stream) {
  (void)in_sizes; (void)n_in; (void)out_size; (void)ws_size;
  const float* h  = (const float*)d_in[0];
  // d_in[1] = causal mask (unused; causality applied analytically)
  const float* Wl = (const float*)d_in[2];
  const float* Wg = (const float*)d_in[3];
  const float* Wv = (const float*)d_in[4];
  const float* Wo = (const float*)d_in[5];
  float* out = (float*)d_out;

  char* ws = (char*)d_ws;
  const size_t SZF  = (size_t)BB * NN * DD * sizeof(float);   // 8 MB
  const size_t SZH  = (size_t)BB * NN * DD * sizeof(bf16);    // 4 MB
  const size_t SZW  = (size_t)DD * DD * sizeof(bf16);         // 128 KB
  size_t off = 0;
  float* zl  = (float*)(ws + off); off += SZF;
  float* zg  = (float*)(ws + off); off += SZF;
  float* v   = (float*)(ws + off); off += SZF;
  float* u   = (float*)(ws + off); off += SZF;
  bf16* znl  = (bf16*)(ws + off);  off += SZH;
  bf16* zgb  = (bf16*)(ws + off);  off += SZH;
  bf16* vT   = (bf16*)(ws + off);  off += SZH;
  bf16* WlT  = (bf16*)(ws + off);  off += SZW;
  bf16* WgT  = (bf16*)(ws + off);  off += SZW;
  bf16* WvT  = (bf16*)(ws + off);  off += SZW;
  bf16* WoT  = (bf16*)(ws + off);  off += SZW;
  float* sq  = (float*)(ws + off); off += (size_t)BB * NN * sizeof(float);

  // K0: weights -> transposed bf16
  k_transpose_w<<<DD, DD, 0, stream>>>(Wl, WlT);
  k_transpose_w<<<DD, DD, 0, stream>>>(Wg, WgT);
  k_transpose_w<<<DD, DD, 0, stream>>>(Wv, WvT);
  k_transpose_w<<<DD, DD, 0, stream>>>(Wo, WoT);

  // K1: projections  ([B*N,256] x [256,256])
  const int M = BB * NN;
  dim3 ggrid(M / 16, 2), gblk(256);
  k_gemm<<<ggrid, gblk, 0, stream>>>(h, WlT, zl);
  k_gemm<<<ggrid, gblk, 0, stream>>>(h, WgT, zg);
  k_gemm<<<ggrid, gblk, 0, stream>>>(h, WvT, v);

  // K2: normalization / squared norms / v transpose
  k_prep<<<BB * NN, DD, 0, stream>>>(zl, zg, v, znl, zgb, vT, sq);

  // K3: fused dual-Laplacian mixing, u = K @ v (wave-pair split over keys)
  k_flash<<<(BB * NN / 16) / 4, 256, 0, stream>>>(znl, zgb, vT, sq, u);

  // K4: out = u @ Wo
  k_gemm<<<ggrid, gblk, 0, stream>>>(u, WoT, out);
}
